// GatingNetwork_9517647527954
// MI455X (gfx1250) — matrix operand order repaired and checked by reading, not compiled, not run on hardware
//
#include <hip/hip_runtime.h>
#include <hip/hip_bf16.h>

typedef __attribute__((ext_vector_type(16))) _Float16 v16h;
typedef __attribute__((ext_vector_type(8)))  float    v8f;

#define D_IN 1024
#define H1   256
#define H2   128
#define NE   64
#define TOPK 8
#define BM   32          // rows per workgroup
// LDS strides (padded to dodge bank conflicts)
#define S_H0 264         // f32 elems per row, h0/h1 tile
#define S_AL 264         // f16 elems per row, LN activation tile
#define S_A3 136         // f16 elems per row, h2 tile
#define S_LG 68          // f32 elems per row, logits tile (reuses h0 region)

// ---------------------------------------------------------------------------
// Pack f32 weight matrix [K][N] (row-major) into f16 WMMA B-fragment layout:
// out[((kt*(N/16)+nt)*32 + lane)*16 + j] = W[kt*32 + (lane>>4)*8 + (j&7) + (j>>3)*16][nt*16 + (lane&15)]
// ---------------------------------------------------------------------------
__global__ void pack_w_kernel(const float* __restrict__ W, _Float16* __restrict__ out,
                              int K, int N) {
  int tid = blockIdx.x * 256 + threadIdx.x;
  if (tid >= K * N) return;
  int j    = tid & 15;
  int lane = (tid >> 4) & 31;
  int frag = tid >> 9;
  int ntiles = N >> 4;
  int nt = frag % ntiles;
  int kt = frag / ntiles;
  int n = nt * 16 + (lane & 15);
  int k = kt * 32 + (lane >> 4) * 8 + (j & 7) + (j >> 3) * 16;
  out[tid] = (_Float16)W[k * N + n];
}

// ---------------------------------------------------------------------------
// Fragment loaders
// ---------------------------------------------------------------------------
// A fragment from global f32 (row-major, ld=1024). p points at row r, col k0+ksub.
__device__ __forceinline__ v16h load_a_f32(const float* __restrict__ p) {
  float4 f0 = *(const float4*)(p);
  float4 f1 = *(const float4*)(p + 4);
  float4 f2 = *(const float4*)(p + 16);
  float4 f3 = *(const float4*)(p + 20);
  v16h h;
  h[0]=(_Float16)f0.x; h[1]=(_Float16)f0.y; h[2]=(_Float16)f0.z; h[3]=(_Float16)f0.w;
  h[4]=(_Float16)f1.x; h[5]=(_Float16)f1.y; h[6]=(_Float16)f1.z; h[7]=(_Float16)f1.w;
  h[8]=(_Float16)f2.x; h[9]=(_Float16)f2.y; h[10]=(_Float16)f2.z; h[11]=(_Float16)f2.w;
  h[12]=(_Float16)f3.x; h[13]=(_Float16)f3.y; h[14]=(_Float16)f3.z; h[15]=(_Float16)f3.w;
  return h;
}

// A fragment from LDS f16. p points at row r, col (k0+ksub). Upper 8 halves at +16.
__device__ __forceinline__ v16h load_a_lds(const _Float16* p) {
  union { v16h h; uint4 u[2]; } r;
  r.u[0] = *(const uint4*)(p);
  r.u[1] = *(const uint4*)(p + 16);
  return r.h;
}

// B fragment from pre-packed global f16 (32 contiguous bytes per lane).
__device__ __forceinline__ v16h load_b_pack(const _Float16* __restrict__ p) {
  union { v16h h; uint4 u[2]; } r;
  r.u[0] = *(const uint4*)(p);
  r.u[1] = *(const uint4*)(p + 8);
  return r.h;
}

// ---------------------------------------------------------------------------
// In-LDS LayerNorm (+ReLU) over 256 cols of a 32-row f32 tile -> f16 tile.
// ---------------------------------------------------------------------------
__device__ __forceinline__ void ln_relu_256(const float* __restrict__ src, _Float16* dst,
                                            const float* __restrict__ g,
                                            const float* __restrict__ bb,
                                            float* SSUM, float* SSQ,
                                            float* SMEAN, float* SRSTD) {
  const int tid = threadIdx.x;
  const int row = tid >> 3, sub = tid & 7;
  const float* rp = src + row * S_H0 + sub * 32;
  float s = 0.f, q = 0.f;
#pragma unroll
  for (int c = 0; c < 32; ++c) { float v = rp[c]; s += v; q += v * v; }
  SSUM[row * 8 + sub] = s; SSQ[row * 8 + sub] = q;
  __syncthreads();
  if (sub == 0) {
    float S = 0.f, Q = 0.f;
#pragma unroll
    for (int i = 0; i < 8; ++i) { S += SSUM[row * 8 + i]; Q += SSQ[row * 8 + i]; }
    float mu  = S * (1.f / 256.f);
    float var = Q * (1.f / 256.f) - mu * mu;
    SMEAN[row] = mu;
    SRSTD[row] = rsqrtf(var + 1e-5f);
  }
  __syncthreads();
  float mu = SMEAN[row], rs = SRSTD[row];
  _Float16* dp = dst + row * S_AL + sub * 32;
  const float* gp = g + sub * 32; const float* bp = bb + sub * 32;
#pragma unroll
  for (int c = 0; c < 32; ++c) {
    float y = (rp[c] - mu) * rs * gp[c] + bp[c];
    dp[c] = (_Float16)fmaxf(y, 0.f);
  }
  __syncthreads();
}

// ---------------------------------------------------------------------------
// Fused gating network: GEMM1 -> LN1 -> GEMM2(+res) -> LN2 -> GEMM3 -> GEMM4
// -> softmax -> top-8.   One workgroup = 32 rows, 8 wave32s, 256 threads.
// ---------------------------------------------------------------------------
__global__ __launch_bounds__(256)
void gating_fused_kernel(const float* __restrict__ x,
                         const _Float16* __restrict__ WpIn,
                         const float* __restrict__ b_in,
                         const float* __restrict__ ln1_g, const float* __restrict__ ln1_b,
                         const _Float16* __restrict__ Wp1,
                         const float* __restrict__ b1,
                         const float* __restrict__ ln2_g, const float* __restrict__ ln2_b,
                         const _Float16* __restrict__ Wp2,
                         const float* __restrict__ b2,
                         const _Float16* __restrict__ WpO,
                         const float* __restrict__ b_out,
                         const float* __restrict__ temperature,
                         float* __restrict__ out_topk,
                         int*   __restrict__ out_idx,
                         float* __restrict__ out_probs) {
  __shared__ __align__(16) unsigned char smem[61696];
  float*    H0    = (float*)smem;                       // 32 x 264 f32 (h0, then h1, then logits)
  _Float16* ALN   = (_Float16*)(smem + 33792);          // 32 x 264 f16 (LN activations)
  _Float16* A3    = (_Float16*)(smem + 50688);          // 32 x 136 f16 (h2)
  float*    SSUM  = (float*)(smem + 59392);
  float*    SSQ   = (float*)(smem + 60416);
  float*    SMEAN = (float*)(smem + 61440);
  float*    SRSTD = (float*)(smem + 61568);
  float*    LOGT  = (float*)smem;                        // reuse H0 region for logits

  const int lane   = threadIdx.x & 31;
  const int wv     = threadIdx.x >> 5;
  const int rstrip = wv >> 2;          // 0..1 : 16-row strip
  const int cg     = wv & 3;           // 0..3 : column group
  const int r      = lane & 15;
  const int hi     = lane >> 4;        // 0/1
  const int ksub   = hi * 8;           // A-frag K sub-offset
  const int rowL   = rstrip * 16;      // wave's local row base
  const int rAcc   = rowL + 8 * hi;    // local row base for acc element v
  const int cn     = lane & 15;        // N within a 16-tile

  // ---------------- GEMM1: h0 = relu(x @ W_in + b_in), K=1024, N=256 -------
  v8f acc[4];
#pragma unroll
  for (int t = 0; t < 4; ++t) acc[t] = (v8f)(0.f);

  const size_t growA = (size_t)blockIdx.x * BM + rowL + r;
  const float* xrow  = x + growA * D_IN;
#pragma unroll 2
  for (int k0 = 0; k0 < D_IN; k0 += 32) {
    const float* ap = xrow + k0 + ksub;
    __builtin_prefetch(ap + 32, 0, 1);            // global_prefetch_b8 for next K block
    v16h a = load_a_f32(ap);
#pragma unroll
    for (int t = 0; t < 4; ++t) {
      int nt = cg * 4 + t;
      v16h b = load_b_pack(WpIn + (((size_t)(k0 >> 5) * (H1 / 16) + nt) * 32 + lane) * 16);
      acc[t] = __builtin_amdgcn_wmma_f32_16x16x32_f16(false, a, false, b,
                                                      (short)0, acc[t], false, false);
    }
  }
#pragma unroll
  for (int t = 0; t < 4; ++t) {
    int col = cg * 64 + t * 16 + cn;
    float bi = b_in[col];
#pragma unroll
    for (int v = 0; v < 8; ++v)
      H0[(rAcc + v) * S_H0 + col] = fmaxf(acc[t][v] + bi, 0.f);
  }
  __syncthreads();

  // ---------------- LN1 + relu -> ALN (f16) --------------------------------
  ln_relu_256(H0, ALN, ln1_g, ln1_b, SSUM, SSQ, SMEAN, SRSTD);

  // ---------------- GEMM2: h1 = ALN @ W1 + b1 + h0, K=256, N=256 -----------
#pragma unroll
  for (int t = 0; t < 4; ++t) acc[t] = (v8f)(0.f);
  const _Float16* aRow2 = ALN + (rowL + r) * S_AL;
#pragma unroll
  for (int k0 = 0; k0 < H1; k0 += 32) {
    v16h a = load_a_lds(aRow2 + k0 + ksub);
#pragma unroll
    for (int t = 0; t < 4; ++t) {
      int nt = cg * 4 + t;
      v16h b = load_b_pack(Wp1 + (((size_t)(k0 >> 5) * (H1 / 16) + nt) * 32 + lane) * 16);
      acc[t] = __builtin_amdgcn_wmma_f32_16x16x32_f16(false, a, false, b,
                                                      (short)0, acc[t], false, false);
    }
  }
#pragma unroll
  for (int t = 0; t < 4; ++t) {
    int col = cg * 64 + t * 16 + cn;
    float bi = b1[col];
#pragma unroll
    for (int v = 0; v < 8; ++v) {
      int off = (rAcc + v) * S_H0 + col;
      H0[off] = acc[t][v] + bi + H0[off];          // residual; in-place (slot owned by lane)
    }
  }
  __syncthreads();

  // ---------------- LN2 + relu -> ALN (f16) --------------------------------
  ln_relu_256(H0, ALN, ln2_g, ln2_b, SSUM, SSQ, SMEAN, SRSTD);

  // ---------------- GEMM3: h2 = ALN @ W2 + b2, K=256, N=128 ----------------
  v8f acc3[2];
  acc3[0] = (v8f)(0.f); acc3[1] = (v8f)(0.f);
#pragma unroll
  for (int k0 = 0; k0 < H1; k0 += 32) {
    v16h a = load_a_lds(aRow2 + k0 + ksub);
#pragma unroll
    for (int t = 0; t < 2; ++t) {
      int nt = cg * 2 + t;
      v16h b = load_b_pack(Wp2 + (((size_t)(k0 >> 5) * (H2 / 16) + nt) * 32 + lane) * 16);
      acc3[t] = __builtin_amdgcn_wmma_f32_16x16x32_f16(false, a, false, b,
                                                       (short)0, acc3[t], false, false);
    }
  }
#pragma unroll
  for (int t = 0; t < 2; ++t) {
    int col = cg * 32 + t * 16 + cn;
    float bi = b2[col];
#pragma unroll
    for (int v = 0; v < 8; ++v)
      A3[(rAcc + v) * S_A3 + col] = (_Float16)(acc3[t][v] + bi);
  }
  __syncthreads();

  // ---------------- GEMM4: logits = h2 @ W_out + b_out, K=128, N=64 --------
  v8f acc4 = (v8f)(0.f);
  const _Float16* aRow4 = A3 + (rowL + r) * S_A3;
#pragma unroll
  for (int k0 = 0; k0 < H2; k0 += 32) {
    v16h a = load_a_lds(aRow4 + k0 + ksub);
    v16h b = load_b_pack(WpO + (((size_t)(k0 >> 5) * (NE / 16) + cg) * 32 + lane) * 16);
    acc4 = __builtin_amdgcn_wmma_f32_16x16x32_f16(false, a, false, b,
                                                  (short)0, acc4, false, false);
  }
  {
    float invT = 1.0f / temperature[0];
    int col = cg * 16 + cn;
    float bo = b_out[col];
#pragma unroll
    for (int v = 0; v < 8; ++v)
      LOGT[(rAcc + v) * S_LG + col] = (acc4[v] + bo) * invT;
  }
  __syncthreads();

  // ---------------- softmax + top-8 (one thread per row) -------------------
  if (threadIdx.x < BM) {
    const int row = threadIdx.x;
    const size_t grow = (size_t)blockIdx.x * BM + row;
    float l[NE];
    float m = -3.4e38f;
#pragma unroll
    for (int c = 0; c < NE; ++c) { l[c] = LOGT[row * S_LG + c]; m = fmaxf(m, l[c]); }
    float s = 0.f;
#pragma unroll
    for (int c = 0; c < NE; ++c) { l[c] = __expf(l[c] - m); s += l[c]; }
    float inv = 1.f / s;
#pragma unroll
    for (int c = 0; c < NE; ++c) l[c] *= inv;
    // gate_probs
    float* op = out_probs + grow * NE;
#pragma unroll
    for (int c = 0; c < NE; c += 4)
      *(float4*)(op + c) = make_float4(l[c], l[c + 1], l[c + 2], l[c + 3]);
    // top-8 by repeated argmax (ties -> earliest index, matching jax top_k)
    float tk[TOPK]; int ti[TOPK]; float tsum = 0.f;
#pragma unroll
    for (int i = 0; i < TOPK; ++i) {
      int bi = 0; float bv = l[0];
      for (int c = 1; c < NE; ++c) if (l[c] > bv) { bv = l[c]; bi = c; }
      tk[i] = bv; ti[i] = bi; tsum += bv; l[bi] = -1.f;
    }
    float it = 1.f / tsum;
#pragma unroll
    for (int i = 0; i < TOPK; ++i) {
      out_topk[grow * TOPK + i] = tk[i] * it;
      out_idx[grow * TOPK + i]  = ti[i];
    }
  }
}

// ---------------------------------------------------------------------------
extern "C" void kernel_launch(void* const* d_in, const int* in_sizes, int n_in,
                              void* d_out, int out_size, void* d_ws, size_t ws_size,
                              hipStream_t stream) {
  const float* x      = (const float*)d_in[0];
  const float* W_in   = (const float*)d_in[1];
  const float* b_in   = (const float*)d_in[2];
  const float* ln1_g  = (const float*)d_in[3];
  const float* ln1_b  = (const float*)d_in[4];
  const float* W1     = (const float*)d_in[5];
  const float* b1     = (const float*)d_in[6];
  const float* ln2_g  = (const float*)d_in[7];
  const float* ln2_b  = (const float*)d_in[8];
  const float* W2     = (const float*)d_in[9];
  const float* b2     = (const float*)d_in[10];
  const float* W_out  = (const float*)d_in[11];
  const float* b_out  = (const float*)d_in[12];
  const float* temp   = (const float*)d_in[13];

  const int B = in_sizes[0] / D_IN;   // 65536

  // workspace: packed f16 weights
  _Float16* ws   = (_Float16*)d_ws;
  _Float16* WpIn = ws;                 // 1024*256 = 262144 halves
  _Float16* Wp1  = ws + 262144;        // 256*256  =  65536
  _Float16* Wp2  = ws + 327680;        // 256*128  =  32768
  _Float16* WpO  = ws + 360448;        // 128*64   =   8192

  pack_w_kernel<<<(D_IN * H1 + 255) / 256, 256, 0, stream>>>(W_in, WpIn, D_IN, H1);
  pack_w_kernel<<<(H1 * H1 + 255) / 256, 256, 0, stream>>>(W1, Wp1, H1, H1);
  pack_w_kernel<<<(H1 * H2 + 255) / 256, 256, 0, stream>>>(W2, Wp2, H1, H2);
  pack_w_kernel<<<(H2 * NE + 255) / 256, 256, 0, stream>>>(W_out, WpO, H2, NE);

  float* out_topk  = (float*)d_out;                       // [B,8]
  int*   out_idx   = (int*)((float*)d_out + (size_t)B * TOPK);   // [B,8] int32 bits
  float* out_probs = (float*)d_out + (size_t)B * 2 * TOPK;       // [B,64]

  gating_fused_kernel<<<B / BM, 256, 0, stream>>>(
      x, WpIn, b_in, ln1_g, ln1_b, Wp1, b1, ln2_g, ln2_b,
      Wp2, b2, WpO, b_out, temp, out_topk, out_idx, out_probs);
}
